// MultiheadLatentAttention_15307263443479
// MI455X (gfx1250) — matrix operand order repaired and checked by reading, not compiled, not run on hardware
//
#include <hip/hip_runtime.h>
#include <math.h>

// ---------------- problem constants (from reference) ----------------
#define D_MODEL  3072
#define N_HEADS  24
#define D_HEAD   128
#define N_KV     6
#define LATENT   512
#define KV_OUT   1536     // 768 K + 768 V
#define SWH      256
#define SOFTCAPF 20.0f
#define NEGINF  -1e30f
#define INV_SQRT_DHEAD 0.08838834764831845f  // 1/sqrt(128)

// ---------------- CDNA5 WMMA types ----------------
typedef __attribute__((ext_vector_type(16))) __bf16 v16bf;
typedef __attribute__((ext_vector_type(8)))  float  v8f;
typedef __attribute__((ext_vector_type(4)))  unsigned int u32x4;
typedef __attribute__((ext_vector_type(8)))  int i32x8;
typedef __attribute__((ext_vector_type(4)))  int i32x4;

union BF16x16 {
  v16bf v;
  unsigned short u[16];
  unsigned int   w[8];
  uint4          q[2];
};

#if defined(__has_builtin)
#if __has_builtin(__builtin_amdgcn_tensor_load_to_lds)
#define HAVE_TDM 1
#endif
#endif
#ifndef HAVE_TDM
#define HAVE_TDM 0
#endif

__device__ __forceinline__ unsigned short f32_to_bf16(float f) {
  unsigned int u = __float_as_uint(f);
  u += 0x7FFFu + ((u >> 16) & 1u);           // round-to-nearest-even
  return (unsigned short)(u >> 16);
}

__device__ __forceinline__ float halfmax16(float v) {
#pragma unroll
  for (int off = 1; off < 16; off <<= 1) v = fmaxf(v, __shfl_xor(v, off, 32));
  return v;  // stays within each 16-lane half of the wave32
}
__device__ __forceinline__ float halfsum16(float v) {
#pragma unroll
  for (int off = 1; off < 16; off <<= 1) v += __shfl_xor(v, off, 32);
  return v;
}

// ---------------- TDM: stage one K tile (32 keys x 128 d, bf16) into LDS ----------
// D# per CDNA5 ISA: group0 = {count|lds_addr|global_addr|type},
// group1 = {data_size, tensor dims, tile dims, tensor_dim0_stride}.
__device__ __forceinline__ void tdm_stage_k(const unsigned short* gsrc,
                                            unsigned short* lds_dst, int lane) {
#if HAVE_TDM
  unsigned long long ga = (unsigned long long)gsrc;
  unsigned lds_off = (unsigned)(unsigned long long)(size_t)lds_dst; // flat addr low 32 = LDS offset
  u32x4 g0;
  g0[0] = 1u;                                           // count=1, user mode
  g0[1] = lds_off;                                      // lds_addr (bytes)
  g0[2] = (unsigned)(ga & 0xFFFFFFFFu);                 // global_addr[31:0]
  g0[3] = (unsigned)((ga >> 32) & 0x01FFFFFFu) | (2u << 30); // global_addr[56:32] | type=2
  i32x8 g1;
  g1[0] = (int)(1u << 16);                              // data_size=1 (2 bytes)
  g1[1] = (int)((unsigned)(KV_OUT & 0xFFFF) << 16);     // tensor_dim0[15:0] = 1536
  g1[2] = (int)(((unsigned)2048 & 0xFFFF) << 16);       // tensor_dim0[31:16]=0 | tensor_dim1[15:0]=2048
  g1[3] = (int)((unsigned)D_HEAD << 16);                // tensor_dim1[31:16]=0 | tile_dim0=128
  g1[4] = 32;                                           // tile_dim1=32 | tile_dim2=0
  g1[5] = KV_OUT;                                       // tensor_dim0_stride[31:0] = 1536
  g1[6] = 0;                                            // stride0 hi | stride1 lo
  g1[7] = 0;
  i32x4 z4 = {0, 0, 0, 0};
#if defined(__clang_major__) && (__clang_major__ >= 23)
  i32x8 z8 = {0, 0, 0, 0, 0, 0, 0, 0};
  __builtin_amdgcn_tensor_load_to_lds(g0, g1, z4, z4, z8, 0);
#else
  __builtin_amdgcn_tensor_load_to_lds(g0, g1, z4, z4, 0);
#endif
#else
  // fallback: cooperative copy, lane = row (key), 8 x 16B
  const uint4* src = (const uint4*)(gsrc + (size_t)lane * KV_OUT);
  uint4* dst = (uint4*)(lds_dst + (size_t)lane * D_HEAD);
#pragma unroll
  for (int i = 0; i < 8; i++) dst[i] = src[i];
#endif
}

__device__ __forceinline__ void tdm_wait_prev() {
#if HAVE_TDM
  __builtin_amdgcn_s_wait_tensorcnt(1);   // previous tile done, next may be in flight
#endif
}
__device__ __forceinline__ void tdm_wait_all() {
#if HAVE_TDM
  __builtin_amdgcn_s_wait_tensorcnt(0);
#endif
}

// ---------------- elementwise f32 -> bf16 ----------------
__global__ void conv_bf16_kernel(const float* __restrict__ src,
                                 unsigned short* __restrict__ dst, int n) {
  int i = blockIdx.x * 256 + threadIdx.x;
  if (i < n) dst[i] = f32_to_bf16(src[i]);
}

// ---------------- transpose + convert: W[K][N] f32 -> WT[N][K] bf16 ----------------
__global__ void convT_kernel(const float* __restrict__ W,
                             unsigned short* __restrict__ WT, int K, int N) {
  __shared__ unsigned short t[32][33];
  int n0 = blockIdx.x * 32, k0 = blockIdx.y * 32;
#pragma unroll
  for (int i = 0; i < 4; i++) {
    int k = k0 + threadIdx.y + i * 8;
    t[threadIdx.y + i * 8][threadIdx.x] =
        f32_to_bf16(W[(size_t)k * N + n0 + threadIdx.x]);
  }
  __syncthreads();
#pragma unroll
  for (int i = 0; i < 4; i++) {
    int n = n0 + threadIdx.y + i * 8;
    WT[(size_t)n * K + k0 + threadIdx.x] = t[threadIdx.x][threadIdx.y + i * 8];
  }
}

// ---------------- V transpose: kvd[(b,t)][768+kvh*128+d] -> vT[b][kvh][d][t] ----------------
__global__ void transpose_v_kernel(const unsigned short* __restrict__ kvd,
                                   unsigned short* __restrict__ vT, int B, int L) {
  long i = (long)blockIdx.x * 256 + threadIdx.x;
  long total = (long)B * L * 768;
  if (i >= total) return;
  int  d0 = (int)(i % 768);
  long bt = i / 768;
  int  t  = (int)(bt % L);
  int  b  = (int)(bt / L);
  int  kvh = d0 >> 7, d = d0 & 127;
  vT[(((size_t)b * N_KV + kvh) * D_HEAD + d) * L + t] = kvd[bt * KV_OUT + 768 + d0];
}

// ---------------- generic bf16 WMMA GEMM ----------------
// C[M,N] = A[M,K] (bf16, row-major) * B (bf16, provided transposed: BT[N,K])
// 256 threads = 8 waves, wave layout 4(M) x 2(N) -> 64x32 block tile.
template <bool OUT_BF16>
__global__ void gemm_wmma_bf16(const unsigned short* __restrict__ A,
                               const unsigned short* __restrict__ BT,
                               float* __restrict__ Cf,
                               unsigned short* __restrict__ Cb,
                               int M, int N, int K) {
  const int lane = threadIdx.x & 31;
  const int wave = threadIdx.x >> 5;
  const int wm = wave & 3, wn = wave >> 2;
  const int tileM = blockIdx.y * 64 + wm * 16;
  const int tileN = blockIdx.x * 32 + wn * 16;
  const int g  = lane >> 4;
  const int ln = lane & 15;

  const unsigned short* arow = A  + (size_t)(tileM + ln) * K;
  const unsigned short* brow = BT + (size_t)(tileN + ln) * K;

  v8f acc = {};
  for (int k0 = 0; k0 < K; k0 += 32) {
    __builtin_prefetch(arow + k0 + 256, 0, 0);
    __builtin_prefetch(brow + k0 + 256, 0, 0);
    BF16x16 af, bf;
    af.q[0] = *(const uint4*)(arow + k0 + g * 8);
    af.q[1] = *(const uint4*)(arow + k0 + 16 + g * 8);
    bf.q[0] = *(const uint4*)(brow + k0 + g * 8);
    bf.q[1] = *(const uint4*)(brow + k0 + 16 + g * 8);
    acc = __builtin_amdgcn_wmma_f32_16x16x32_bf16(false, af.v, false, bf.v,
                                                  (short)0, acc, false, false);
  }
#pragma unroll
  for (int r = 0; r < 8; r++) {
    int row = tileM + r + 8 * g;
    size_t idx = (size_t)row * N + tileN + ln;
    if (OUT_BF16) Cb[idx] = f32_to_bf16(acc[r]);
    else          Cf[idx] = acc[r];
  }
}

// ---------------- fused flash attention (short sliding-window + long full) ----------------
// grid: (L/16 query tiles, 24 heads, B). block: 32 threads (one wave).
// K tiles are streamed into LDS by the Tensor Data Mover, double-buffered.
__global__ void mla_attn_kernel(const unsigned short* __restrict__ qb,   // [B][L][24][128]
                                const unsigned short* __restrict__ kvd,  // [B][L][1536] (K at 0..767)
                                const unsigned short* __restrict__ vT,   // [B][6][128][L]
                                unsigned short* __restrict__ outb,       // [B][L][24][128]
                                const float* __restrict__ sscale,
                                int B, int L) {
  __shared__ __align__(256) unsigned short kst[2][32][128];  // double-buffered K tile
  __shared__ unsigned short lds_p[16][34];                   // P repack tile

  const int lane = threadIdx.x & 31;
  const int g = lane >> 4, ln = lane & 15;
  const int qt = blockIdx.x, h = blockIdx.y, b = blockIdx.z;
  const int q0 = qt * 16;
  const bool windowed = (h < 16);
  const int kvh = windowed ? (h >> 2) : 4 + ((h - 16) >> 2);
  const float comb = sscale[h] * logf((float)L) * INV_SQRT_DHEAD;

  // ---- Q fragments: single base pointer, immediate offsets ----
  BF16x16 qf[4];
  {
    const unsigned short* qrow =
        qb + (unsigned)(((unsigned)(b * L + q0 + ln) * N_HEADS + h) * D_HEAD);
#pragma unroll
    for (int c = 0; c < 4; c++) {
      qf[c].q[0] = *(const uint4*)(qrow + c * 32 + g * 16);
      qf[c].q[1] = *(const uint4*)(qrow + c * 32 + g * 16 + 8);
    }
  }

  v8f zero = {};
  v8f o[8];
#pragma unroll
  for (int i = 0; i < 8; i++) o[i] = zero;
  float mrun[8], lrun[8];
#pragma unroll
  for (int r = 0; r < 8; r++) { mrun[r] = NEGINF; lrun[r] = 0.0f; }

  const int nchunks = windowed ? 17 : (L >> 5);
  const int base    = windowed ? (q0 - SWH) : 0;

  // uniform K-tile source (row 0, this batch + kv head)
  const unsigned short* ksrc0 = kvd + (size_t)b * L * KV_OUT + kvh * D_HEAD;
  // uniform V base for this lane's column; d-tile offset folded into imm offsets
  const unsigned short* vlane =
      vT + (unsigned)((unsigned)(b * N_KV + kvh) * (unsigned)(D_HEAD * L)) +
      (unsigned)ln * (unsigned)L;

  // prologue: stage chunk 0
  {
    int n0c = min(max(base, 0), L - 32);
    tdm_stage_k(ksrc0 + (size_t)n0c * KV_OUT, &kst[0][0][0], lane);
  }

  for (int ci = 0; ci < nchunks; ci++) {
    const int cur = ci & 1;
    const int n0  = base + ci * 32;
    const int n0c = min(max(n0, 0), L - 32);

    if (ci + 1 < nchunks) {  // stage next tile while we compute
      int n1c = min(max(n0 + 32, 0), L - 32);
      tdm_stage_k(ksrc0 + (size_t)n1c * KV_OUT, &kst[cur ^ 1][0][0], lane);
      tdm_wait_prev();
    } else {
      tdm_wait_all();
    }

    // ---- S = Q K^T (16 queries x 32 keys), K fragments from LDS ----
    v8f s0 = {}, s1 = {};
    {
      int loc0 = min(max(n0 + ln - n0c, 0), 31);       // LDS row for s0 keys
      int loc1 = min(max(n0 + 16 + ln - n0c, 0), 31);  // LDS row for s1 keys
      const unsigned short* kr0 = &kst[cur][loc0][0];
      const unsigned short* kr1 = &kst[cur][loc1][0];
#pragma unroll
      for (int c = 0; c < 4; c++) {
        BF16x16 kf0, kf1;
        kf0.q[0] = *(const uint4*)(kr0 + c * 32 + g * 16);
        kf0.q[1] = *(const uint4*)(kr0 + c * 32 + g * 16 + 8);
        kf1.q[0] = *(const uint4*)(kr1 + c * 32 + g * 16);
        kf1.q[1] = *(const uint4*)(kr1 + c * 32 + g * 16 + 8);
        s0 = __builtin_amdgcn_wmma_f32_16x16x32_bf16(false, qf[c].v, false, kf0.v,
                                                     (short)0, s0, false, false);
        s1 = __builtin_amdgcn_wmma_f32_16x16x32_bf16(false, qf[c].v, false, kf1.v,
                                                     (short)0, s1, false, false);
      }
    }

    // ---- softcap + mask + online softmax update ----
#pragma unroll
    for (int r = 0; r < 8; r++) {
      const int i_row = q0 + r + 8 * g;
      const int j0 = n0 + ln, j1 = n0 + 16 + ln;

      float x0 = SOFTCAPF * tanhf(s0[r] * comb * (1.0f / SOFTCAPF));
      float x1 = SOFTCAPF * tanhf(s1[r] * comb * (1.0f / SOFTCAPF));
      int dd0 = i_row - j0, dd1 = i_row - j1;
      bool ok0 = (j0 >= 0) && (j0 < L) && (!windowed || (dd0 <= SWH && dd0 >= -SWH));
      bool ok1 = (j1 >= 0) && (j1 < L) && (!windowed || (dd1 <= SWH && dd1 >= -SWH));
      x0 = ok0 ? x0 : NEGINF;
      x1 = ok1 ? x1 : NEGINF;

      float mx = halfmax16(fmaxf(x0, x1));
      float mnew = fmaxf(mrun[r], mx);
      float alpha = __expf(mrun[r] - mnew);
      float p0 = __expf(x0 - mnew);
      float p1 = __expf(x1 - mnew);
      lrun[r] = lrun[r] * alpha + halfsum16(p0 + p1);
      mrun[r] = mnew;
#pragma unroll
      for (int nt = 0; nt < 8; nt++) o[nt][r] *= alpha;

      lds_p[r + 8 * g][ln]      = f32_to_bf16(p0);
      lds_p[r + 8 * g][16 + ln] = f32_to_bf16(p1);
    }

    // ---- reload P as A-fragment ----
    BF16x16 pf;
    {
      const unsigned short* prow = &lds_p[ln][0];
#pragma unroll
      for (int e = 0; e < 8; e++) {
        pf.u[e]     = prow[g * 16 + e];
        pf.u[8 + e] = prow[g * 16 + 8 + e];
      }
    }

    // ---- O += P V : one base pointer, d-tile offset = nt*16*L folded to imm ----
    {
      int kstart = min(max(n0 + g * 16, 0), L - 16);   // OOB keys have P==0
      const unsigned short* vp = vlane + kstart;
#pragma unroll
      for (int nt = 0; nt < 8; nt++) {
        BF16x16 vf;
        vf.q[0] = *(const uint4*)(vp + nt * (16 * 2048));
        vf.q[1] = *(const uint4*)(vp + nt * (16 * 2048) + 8);
        o[nt] = __builtin_amdgcn_wmma_f32_16x16x32_bf16(false, pf.v, false, vf.v,
                                                        (short)0, o[nt], false, false);
      }
    }
  }

  // ---- normalize and store (base pointer + imm offsets) ----
#pragma unroll
  for (int r = 0; r < 8; r++) {
    float inv = 1.0f / lrun[r];
    unsigned short* orow =
        outb + (unsigned)(((unsigned)(b * L + q0 + r + 8 * g) * N_HEADS + h) * D_HEAD) + ln;
#pragma unroll
    for (int nt = 0; nt < 8; nt++)
      orow[nt * 16] = f32_to_bf16(o[nt][r] * inv);
  }
}

// ---------------- host side ----------------
extern "C" void kernel_launch(void* const* d_in, const int* in_sizes, int n_in,
                              void* d_out, int out_size, void* d_ws, size_t ws_size,
                              hipStream_t stream) {
  const float* query  = (const float*)d_in[0];
  const float* kv     = (const float*)d_in[1];
  const float* w_q    = (const float*)d_in[2];
  const float* w_c    = (const float*)d_in[3];
  const float* w_d    = (const float*)d_in[4];
  const float* w_o    = (const float*)d_in[5];
  const float* sscale = (const float*)d_in[6];

  const int BL = in_sizes[0] / D_MODEL;   // B*L
  const int L  = 2048;
  const int B  = BL / L;

  char* p = (char*)d_ws;
  auto take = [&](size_t bytes) {
    void* r = (void*)p;
    p += (bytes + 255) & ~(size_t)255;
    return r;
  };
  unsigned short* wqT   = (unsigned short*)take((size_t)D_MODEL * D_MODEL * 2);
  unsigned short* wcT   = (unsigned short*)take((size_t)LATENT  * D_MODEL * 2);
  unsigned short* wdT   = (unsigned short*)take((size_t)KV_OUT  * LATENT  * 2);
  unsigned short* woT   = (unsigned short*)take((size_t)D_MODEL * D_MODEL * 2);
  unsigned short* q_bf  = (unsigned short*)take((size_t)BL * D_MODEL * 2);
  unsigned short* kv_bf = (unsigned short*)take((size_t)BL * D_MODEL * 2);
  unsigned short* qproj = (unsigned short*)take((size_t)BL * D_MODEL * 2);
  unsigned short* lat   = (unsigned short*)take((size_t)BL * LATENT  * 2);
  unsigned short* kvd   = (unsigned short*)take((size_t)BL * KV_OUT  * 2);
  unsigned short* vT    = (unsigned short*)take((size_t)B * N_KV * D_HEAD * L * 2);
  unsigned short* attn  = (unsigned short*)take((size_t)BL * D_MODEL * 2);

  {
    int n = BL * D_MODEL;
    conv_bf16_kernel<<<(n + 255) / 256, 256, 0, stream>>>(query, q_bf, n);
    conv_bf16_kernel<<<(n + 255) / 256, 256, 0, stream>>>(kv, kv_bf, n);
  }
  {
    dim3 blk(32, 8);
    convT_kernel<<<dim3(D_MODEL / 32, D_MODEL / 32), blk, 0, stream>>>(w_q, wqT, D_MODEL, D_MODEL);
    convT_kernel<<<dim3(LATENT  / 32, D_MODEL / 32), blk, 0, stream>>>(w_c, wcT, D_MODEL, LATENT);
    convT_kernel<<<dim3(KV_OUT  / 32, LATENT  / 32), blk, 0, stream>>>(w_d, wdT, LATENT, KV_OUT);
    convT_kernel<<<dim3(D_MODEL / 32, D_MODEL / 32), blk, 0, stream>>>(w_o, woT, D_MODEL, D_MODEL);
  }
  gemm_wmma_bf16<true><<<dim3(D_MODEL / 32, BL / 64), 256, 0, stream>>>(
      q_bf, wqT, nullptr, qproj, BL, D_MODEL, D_MODEL);
  gemm_wmma_bf16<true><<<dim3(LATENT / 32, BL / 64), 256, 0, stream>>>(
      kv_bf, wcT, nullptr, lat, BL, LATENT, D_MODEL);
  gemm_wmma_bf16<true><<<dim3(KV_OUT / 32, BL / 64), 256, 0, stream>>>(
      lat, wdT, nullptr, kvd, BL, KV_OUT, LATENT);
  {
    long n = (long)B * L * 768;
    transpose_v_kernel<<<(unsigned)((n + 255) / 256), 256, 0, stream>>>(kvd, vT, B, L);
  }
  mla_attn_kernel<<<dim3(L / 16, N_HEADS, B), 32, 0, stream>>>(
      qproj, kvd, vT, attn, sscale, B, L);
  gemm_wmma_bf16<false><<<dim3(D_MODEL / 32, BL / 64), 256, 0, stream>>>(
      attn, woT, (float*)d_out, nullptr, BL, D_MODEL, D_MODEL);
}
